// ScaledDotProductAttention_15049565405878
// MI455X (gfx1250) — matrix operand order repaired and checked by reading, not compiled, not run on hardware
//
#include <hip/hip_runtime.h>

// MI455X / gfx1250 flash-attention-style forward that also materializes the
// full attention matrix (reference returns (context, attn)).
// f16 WMMA with fp32 accumulation: compute is far below the 23.3 TB/s HBM
// floor set by the 256 MB attn write; all inputs (25 MB) live in the 192 MB L2.
// The attn output stream uses non-temporal b128 stores so the 256 MB
// write-once stream does not evict the L2-resident K/V that every q-tile
// re-reads.

typedef __attribute__((ext_vector_type(16))) _Float16 v16h;
typedef __attribute__((ext_vector_type(8)))  float    v8f;
typedef __attribute__((ext_vector_type(4)))  float    v4f;

#define B_    16
#define S_    2048
#define D_    64
#define QT    16          // queries per workgroup
#define SSTR  2052        // padded fp32 row stride for score buffer (2048+4)
#define NWAVE 8           // 256 threads = 8 wave32

__global__ __launch_bounds__(256)
void attn_fwd_kernel(const float* __restrict__ Q,
                     const float* __restrict__ K,
                     const float* __restrict__ V,
                     float* __restrict__ out)
{
    __shared__ float    sS[QT * SSTR];   // scores -> probs -> (reused) ctx partials
    __shared__ _Float16 sQh[QT * D_];    // scaled Q tile, f16
    __shared__ float    sRed[256];
    __shared__ float    sMax[QT];
    __shared__ float    sInv[QT];

    const int t    = threadIdx.x;
    const int lane = t & 31;
    const int w    = t >> 5;
    const int bb   = blockIdx.x >> 7;    // batch index      (16)
    const int qt   = blockIdx.x & 127;   // query-tile index (128)
    const int qbase = qt * QT;

    const int mrow = lane & 15;          // row within 16 (A/B/N index)
    const int hi   = (lane >> 4) & 1;    // lane-half selector

    // ---------------- load + scale (1/sqrt(64)=0.125) + convert Q tile ----------------
    const size_t qBase = ((size_t)bb * S_ + qbase) * D_;
    #pragma unroll
    for (int i = 0; i < 4; ++i) {
        const int idx = t + i * 256;                 // 16*64 = 1024 elements
        sQh[idx] = (_Float16)(Q[qBase + idx] * 0.125f);
    }
    __syncthreads();

    // ---------------- per-wave A fragments of Q (16x32 f16, two K halves) -------------
    // Contiguous k order chosen here (B uses the same): lane<16 holds K in {0..15}+off, lane>=16 holds {16..31}+off
    v16h aq0, aq1;
    #pragma unroll
    for (int j = 0; j < 16; ++j) {
        const int d0 = (hi << 4) + j;
        aq0[j] = sQh[mrow * D_ + d0];
        aq1[j] = sQh[mrow * D_ + 32 + d0];
    }

    // ---------------- phase 1: S = (Q*scale) @ K^T over 128 key tiles -----------------
    for (int kt = w; kt < S_ / 16; kt += NWAVE) {
        const int kbase = kt * 16;
        // B fragment (K^T, 32x16): lane column n = mrow -> key row kbase+mrow,
        // lane half selects K(dim) offset 0/16 within each 32-chunk.
        const size_t krow = ((size_t)bb * S_ + kbase + mrow) * D_;
        v16h b0, b1;
        #pragma unroll
        for (int j = 0; j < 16; ++j) {
            const int d0 = (hi << 4) + j;            // 16 consecutive floats -> b128 loads
            b0[j] = (_Float16)K[krow + d0];
            b1[j] = (_Float16)K[krow + 32 + d0];
        }
        v8f c = {};
        c = __builtin_amdgcn_wmma_f32_16x16x32_f16(false, aq0, false, b0, (short)0, c, false, false);
        c = __builtin_amdgcn_wmma_f32_16x16x32_f16(false, aq1, false, b1, (short)0, c, false, false);
        #pragma unroll
        for (int r = 0; r < 8; ++r) {
            const int m = r + (hi << 3);
            sS[m * SSTR + kbase + mrow] = c[r];
        }
    }
    __syncthreads();

    // ---------------- phase 2: softmax stats + in-place normalize ---------------------
    const int r2 = t & 15;          // row
    const int c2 = t >> 4;          // 128-wide chunk
    float* row = &sS[r2 * SSTR + c2 * 128];

    float pmax = -3.0e38f;
    for (int j = 0; j < 128; ++j) pmax = fmaxf(pmax, row[j]);
    sRed[t] = pmax;
    __syncthreads();
    if (t < 16) {
        float m = -3.0e38f;
        #pragma unroll
        for (int i = 0; i < 16; ++i) m = fmaxf(m, sRed[i * 16 + t]);
        sMax[t] = m;
    }
    __syncthreads();

    const float rmax = sMax[r2];
    float psum = 0.0f;
    for (int j = 0; j < 128; ++j) {
        const float e = __expf(row[j] - rmax);
        row[j] = e;
        psum += e;
    }
    sRed[t] = psum;
    __syncthreads();
    if (t < 16) {
        float s = 0.0f;
        #pragma unroll
        for (int i = 0; i < 16; ++i) s += sRed[i * 16 + t];
        sInv[t] = 1.0f / s;
    }
    __syncthreads();

    const float rinv = sInv[r2];
    for (int j = 0; j < 128; ++j) row[j] *= rinv;
    __syncthreads();

    // ---------------- phase 3a: context partials = P @ V via WMMA ---------------------
    v8f acc[4] = {v8f{}, v8f{}, v8f{}, v8f{}};
    for (int kc = w; kc < S_ / 32; kc += NWAVE) {
        // A fragment: P 16x32 (f16)
        v16h ap;
        #pragma unroll
        for (int j = 0; j < 16; ++j) {
            const int kk = kc * 32 + (hi << 4) + j;
            ap[j] = (_Float16)sS[mrow * SSTR + kk];
        }
        // B fragments: V 32x16 per n-tile; lane column = d = nt*16 + mrow
        const size_t vrow = ((size_t)bb * S_ + (size_t)kc * 32 + (size_t)hi * 16) * D_ + mrow;
        #pragma unroll
        for (int nt = 0; nt < 4; ++nt) {
            v16h bv;
            #pragma unroll
            for (int j = 0; j < 16; ++j)
                bv[j] = (_Float16)V[vrow + nt * 16 + (size_t)j * D_];
            acc[nt] = __builtin_amdgcn_wmma_f32_16x16x32_f16(false, ap, false, bv,
                                                             (short)0, acc[nt], false, false);
        }
    }

    // ---------------- phase 3b: non-temporal b128 writes of attn (16 x 2048) ----------
    // 256 MB stream, written once, never re-read: NT temporal hint keeps the
    // 192 MB L2 holding Q/K/V instead of dead output lines.
    {
        const size_t attnBase = (size_t)B_ * S_ * D_;
        const size_t aRow = attnBase + ((size_t)bb * S_ + qbase) * S_;
        #pragma unroll 4
        for (int i = 0; i < 32; ++i) {
            const int idx4 = i * 256 + t;            // 8192 float4s = 32768 floats
            const int m  = idx4 >> 9;                // 512 float4s per row
            const int k4 = idx4 & 511;
            const float* src = &sS[m * SSTR + (k4 << 2)];   // 16B aligned (8208 = 16*513)
            v4f val = { src[0], src[1], src[2], src[3] };
            __builtin_nontemporal_store(val, (v4f*)(out + aRow + (size_t)m * S_ + (k4 << 2)));
        }
    }
    __syncthreads();   // everyone done reading sS -> safe to reuse as reduction buffer

    // ---------------- phase 3c: cross-wave reduce of context, write out ---------------
    #pragma unroll
    for (int nt = 0; nt < 4; ++nt) {
        #pragma unroll
        for (int r = 0; r < 8; ++r) {
            const int m = r + (hi << 3);
            const int d = nt * 16 + mrow;
            sS[w * (QT * D_) + m * D_ + d] = acc[nt][r];
        }
    }
    __syncthreads();

    const size_t ctxRow = ((size_t)bb * S_ + qbase) * D_;
    #pragma unroll
    for (int i = 0; i < 4; ++i) {
        const int e = t + i * 256;                   // 16*64 = 1024 elements
        float s = 0.0f;
        #pragma unroll
        for (int ww = 0; ww < NWAVE; ++ww) s += sS[ww * (QT * D_) + e];
        out[ctxRow + e] = s;
    }
}

extern "C" void kernel_launch(void* const* d_in, const int* in_sizes, int n_in,
                              void* d_out, int out_size, void* d_ws, size_t ws_size,
                              hipStream_t stream) {
    (void)in_sizes; (void)n_in; (void)out_size; (void)d_ws; (void)ws_size;
    const float* Q = (const float*)d_in[0];
    const float* K = (const float*)d_in[1];
    const float* V = (const float*)d_in[2];
    float* out = (float*)d_out;
    dim3 grid(B_ * (S_ / QT));   // 16 * 128 = 2048 workgroups
    dim3 block(256);             // 8 wave32
    attn_fwd_kernel<<<grid, block, 0, stream>>>(Q, K, V, out);
}
